// MambaMixerBlock_88639535054982
// MI455X (gfx1250) — matrix operand order repaired and checked by reading, not compiled, not run on hardware
//
#include <hip/hip_runtime.h>
#include <hip/hip_bf16.h>
#include <math.h>

// MambaMixerBlock for MI455X (gfx1250, wave32).
// All GEMMs run guard-free on v_wmma_f32_16x16x32_f16: activations/weights are
// pre-packed to padded f16 layouts so the hot loop is 8x global_load_b128 +
// 4x v_wmma per K-step with zero branches (EXEC all-ones, as WMMA requires).
// Selective scans: one wave32 per (batch, head), 128-wide state in 4 VGPRs/lane.

typedef __attribute__((ext_vector_type(16))) _Float16 v16h;
typedef __attribute__((ext_vector_type(8)))  _Float16 v8h;
typedef __attribute__((ext_vector_type(8)))  float    v8f;

#define B_   8
#define P_   196
#define D_   512
#define TH_  128
#define CH_  512
#define NS_  128   // state size n

__device__ __forceinline__ float silu_f(float x)     { return x / (1.0f + __expf(-x)); }
__device__ __forceinline__ float softplus_f(float x) { return x > 20.0f ? x : log1pf(__expf(x)); }

#define WMMA_F16(a, b, c) __builtin_amdgcn_wmma_f32_16x16x32_f16( \
    false, (a), false, (b), (short)0, (c), false, false)

#define CAT16(lo8, hi8) __builtin_shufflevector((lo8), (hi8), \
    0, 1, 2, 3, 4, 5, 6, 7, 8, 9, 10, 11, 12, 13, 14, 15)

// ---------------------------------------------------------------------------
// Guard-free WMMA GEMM: C[M,Npad] = A16[M,Kp] @ Bt16[Npad,Kp]^T (+bias) (+resid)
// A16: row-major f16, lda halves (Kp, multiple of 32).
// Bt16: TRANSPOSED weights, row n holds K-contiguous halves (ldbt = Kp).
// One wave per 32x32 C tile: 2 A-frags x 2 B-frags -> 4 wmma / K-step.
// M and Npad must be multiples of 32 (guaranteed by padded packing).
// ---------------------------------------------------------------------------
__global__ __launch_bounds__(32)
void wmma_gemm_kernel(const _Float16* __restrict__ A, const _Float16* __restrict__ Bt,
                      float* __restrict__ C, const float* __restrict__ bias,
                      const float* __restrict__ resid,
                      int Kp, int lda, int ldbt, int ldc, int Nreal)
{
    const int lane = threadIdx.x;
    const int lo = lane & 15, hi = lane >> 4;
    const int n0 = blockIdx.x * 32;
    const int m0 = blockIdx.y * 32;

    v8f acc00 = {0.f,0.f,0.f,0.f,0.f,0.f,0.f,0.f};
    v8f acc01 = acc00, acc10 = acc00, acc11 = acc00;

    // A fragment (ISA 7.12.2, 16-bit A 16x32): lane row = lo,
    //   e0..7  -> K = k0 + hi*8 + e ;  e8..15 -> K = k0 + 16 + hi*8 + e
    const _Float16* pa0 = A + (long)(m0 + lo) * lda + hi * 8;
    const _Float16* pa1 = pa0 + (long)16 * lda;
    // B fragment: lane col = lo, same K mapping as A (e0..7 -> k0+hi*8+e, e8..15 -> k0+16+hi*8+e)
    const _Float16* pb0 = Bt + (long)(n0 + lo) * ldbt + hi * 8;
    const _Float16* pb1 = pb0 + (long)16 * ldbt;

    for (int k0 = 0; k0 < Kp; k0 += 32) {
        v8h a0l = *(const v8h*)(pa0 + k0);
        v8h a0h = *(const v8h*)(pa0 + k0 + 16);
        v8h a1l = *(const v8h*)(pa1 + k0);
        v8h a1h = *(const v8h*)(pa1 + k0 + 16);
        v8h b0l = *(const v8h*)(pb0 + k0);
        v8h b0h = *(const v8h*)(pb0 + k0 + 16);
        v8h b1l = *(const v8h*)(pb1 + k0);
        v8h b1h = *(const v8h*)(pb1 + k0 + 16);
        if (k0 + 32 < Kp) {
            __builtin_prefetch(pa0 + k0 + 32, 0, 1);   // global_prefetch_b8
            __builtin_prefetch(pb0 + k0 + 32, 0, 1);
        }
        v16h af0 = CAT16(a0l, a0h);
        v16h af1 = CAT16(a1l, a1h);
        v16h bf0 = CAT16(b0l, b0h);
        v16h bf1 = CAT16(b1l, b1h);
        acc00 = WMMA_F16(af0, bf0, acc00);
        acc01 = WMMA_F16(af0, bf1, acc01);
        acc10 = WMMA_F16(af1, bf0, acc10);
        acc11 = WMMA_F16(af1, bf1, acc11);
    }

    // D layout: VGPR r -> row = hi*8 + r, col = lo (per 16x16 tile)
    const int c0 = n0 + lo, c1 = n0 + 16 + lo;
    const float bv0 = (bias && c0 < Nreal) ? bias[c0] : 0.0f;
    const float bv1 = (bias && c1 < Nreal) ? bias[c1] : 0.0f;
#pragma unroll
    for (int r = 0; r < 8; ++r) {
        const int row0 = m0 + hi * 8 + r;
        const int row1 = row0 + 16;
        float v00 = acc00[r] + bv0, v01 = acc01[r] + bv1;
        float v10 = acc10[r] + bv0, v11 = acc11[r] + bv1;
        if (resid) {
            v00 += resid[(long)row0 * ldc + c0];
            v01 += resid[(long)row0 * ldc + c1];
            v10 += resid[(long)row1 * ldc + c0];
            v11 += resid[(long)row1 * ldc + c1];
        }
        C[(long)row0 * ldc + c0] = v00;
        C[(long)row0 * ldc + c1] = v01;
        C[(long)row1 * ldc + c0] = v10;
        C[(long)row1 * ldc + c1] = v11;
    }
}

// ---------------------------------------------------------------------------
// Pack fp32 weight W[K,N] -> transposed padded f16 Wt[Npad, Kp] (row n = K-major)
// Full coverage of Wt (pads written as zero) => no memset needed.
// ---------------------------------------------------------------------------
__global__ void pack_weight_kernel(const float* __restrict__ W, _Float16* __restrict__ Wt,
                                   int K, int N, int Kp, int total)
{
    int idx = blockIdx.x * blockDim.x + threadIdx.x;
    if (idx >= total) return;
    int k = idx % Kp, n = idx / Kp;
    float v = (k < K && n < N) ? W[(long)k * N + n] : 0.0f;
    Wt[idx] = (_Float16)v;
}

// ---------------------------------------------------------------------------
// RMSNorm over last axis (len Dn); writes f16 for the following GEMM.
// transpose=1: out16[(b*Dn+d)*Pp + p] (pad cols must be pre-zeroed)
// transpose=0: out16[row*Dn + d]
// ---------------------------------------------------------------------------
__global__ __launch_bounds__(256)
void rmsnorm_kernel(const float* __restrict__ x, const float* __restrict__ w,
                    _Float16* __restrict__ out, int Dn, int Pn, int Pp, int transpose)
{
    const int row = blockIdx.x;              // b*Pn + p
    const float* xr = x + (long)row * Dn;
    float ss = 0.f;
    for (int d = threadIdx.x; d < Dn; d += blockDim.x) { float v = xr[d]; ss += v * v; }

    __shared__ float red[8];
    for (int off = 16; off > 0; off >>= 1) ss += __shfl_xor(ss, off);
    if ((threadIdx.x & 31) == 0) red[threadIdx.x >> 5] = ss;
    __syncthreads();
    if (threadIdx.x < 8) {
        float v = red[threadIdx.x];
        for (int off = 4; off > 0; off >>= 1) v += __shfl_xor(v, off);
        if (threadIdx.x == 0) red[0] = v;
    }
    __syncthreads();
    const float rs = rsqrtf(red[0] / (float)Dn + 1e-8f);

    const int b = row / Pn, p = row % Pn;
    for (int d = threadIdx.x; d < Dn; d += blockDim.x) {
        float v = xr[d] * rs * w[d];
        if (transpose) out[((long)b * Dn + d) * Pp + p] = (_Float16)v;
        else           out[(long)row * Dn + d] = (_Float16)v;
    }
}

// ---------------------------------------------------------------------------
// Causal depthwise conv (K=2) + bias + SiLU; emits fp32 (scan) and f16 (GEMM).
// ---------------------------------------------------------------------------
__global__ void conv_silu_kernel(const float* __restrict__ xz, const float* __restrict__ convw,
                                 const float* __restrict__ convb, float* __restrict__ xc,
                                 _Float16* __restrict__ xc16,
                                 int L, int H, int xzld, int total)
{
    int idx = blockIdx.x * blockDim.x + threadIdx.x;
    if (idx >= total) return;
    int h = idx % H;
    int l = (idx / H) % L;
    int b = idx / (H * L);
    float cur  = xz[((long)b * L + l) * xzld + h];
    float prev = (l > 0) ? xz[((long)b * L + l - 1) * xzld + h] : 0.0f;
    float v = silu_f(convw[h * 2 + 0] * prev + convw[h * 2 + 1] * cur + convb[h]);
    xc[idx]   = v;
    xc16[idx] = (_Float16)v;
}

// Slice proj[:, :dtr] into padded f16 A (M x 32) for the dt GEMM.
__global__ void dtslice_kernel(const float* __restrict__ proj, _Float16* __restrict__ out,
                               int ldproj, int dtr, int total)
{
    int idx = blockIdx.x * blockDim.x + threadIdx.x;
    if (idx >= total) return;
    int k = idx & 31, m = idx >> 5;
    float v = (k < dtr) ? proj[(long)m * ldproj + k] : 0.0f;
    out[idx] = (_Float16)v;
}

// ---------------------------------------------------------------------------
// Selective scan: one wave32 per (b,h). State (n=128) in 4 VGPRs/lane.
// y16[b,l,h] = f16( (sum_n h_n*C_n + Dp[h]*xc) * silu(z) )
// ---------------------------------------------------------------------------
__global__ __launch_bounds__(32)
void mamba_scan_kernel(const float* __restrict__ xc, const float* __restrict__ proj,
                       const float* __restrict__ dtraw, const float* __restrict__ Alog,
                       const float* __restrict__ Dp, const float* __restrict__ xz,
                       _Float16* __restrict__ yout,
                       int L, int H, int dtr, int projld, int xzld)
{
    const int bh = blockIdx.x;
    const int b = bh / H, h = bh % H;
    const int lane = threadIdx.x;

    float Acoef[4], st[4];
#pragma unroll
    for (int j = 0; j < 4; ++j) {
        Acoef[j] = -__expf(Alog[(long)h * NS_ + lane + 32 * j]);
        st[j] = 0.f;
    }
    const float dph = Dp[h];

    for (int l = 0; l < L; ++l) {
        const long rl = (long)b * L + l;
        const float dt  = softplus_f(dtraw[rl * H + h]);
        const float xcv = xc[rl * H + h];
        const float* pb = proj + rl * projld + dtr;
        const float dtx = dt * xcv;
        float part = 0.f;
#pragma unroll
        for (int j = 0; j < 4; ++j) {
            float Bv = pb[lane + 32 * j];
            float Cv = pb[NS_ + lane + 32 * j];
            st[j] = __expf(dt * Acoef[j]) * st[j] + dtx * Bv;
            part += st[j] * Cv;
        }
        for (int off = 16; off > 0; off >>= 1) part += __shfl_xor(part, off);
        if (lane == 0) {
            float z = xz[rl * xzld + H + h];
            yout[rl * H + h] = (_Float16)((part + dph * xcv) * silu_f(z));
        }
    }
}

// token_mixer[b,p,d] = tm_out[b,d,p] + x[b,p,d]   (tm_out has ldc=224)
__global__ void transpose_add_kernel(const float* __restrict__ tmout, const float* __restrict__ x,
                                     float* __restrict__ out, int total)
{
    int idx = blockIdx.x * blockDim.x + threadIdx.x;
    if (idx >= total) return;
    int d = idx % D_;
    int p = (idx / D_) % P_;
    int b = idx / (D_ * P_);
    out[idx] = tmout[((long)b * D_ + d) * 224 + p] + x[idx];
}

// ---------------------------------------------------------------------------
extern "C" void kernel_launch(void* const* d_in, const int* in_sizes, int n_in,
                              void* d_out, int out_size, void* d_ws, size_t ws_size,
                              hipStream_t stream)
{
    (void)in_sizes; (void)n_in; (void)out_size; (void)ws_size;
    const float* x         = (const float*)d_in[0];
    const float* tm_norm_w = (const float*)d_in[1];
    const float* cm_norm_w = (const float*)d_in[2];
    const float* tm_Win  = (const float*)d_in[3];
    const float* tm_bin  = (const float*)d_in[4];
    const float* tm_convw= (const float*)d_in[5];
    const float* tm_convb= (const float*)d_in[6];
    const float* tm_Wx   = (const float*)d_in[7];
    const float* tm_Wdt  = (const float*)d_in[8];
    const float* tm_bdt  = (const float*)d_in[9];
    const float* tm_Alog = (const float*)d_in[10];
    const float* tm_Dp   = (const float*)d_in[11];
    const float* tm_Wout = (const float*)d_in[12];
    const float* tm_bout = (const float*)d_in[13];
    const float* cm_Win  = (const float*)d_in[14];
    const float* cm_bin  = (const float*)d_in[15];
    const float* cm_convw= (const float*)d_in[16];
    const float* cm_convb= (const float*)d_in[17];
    const float* cm_Wx   = (const float*)d_in[18];
    const float* cm_Wdt  = (const float*)d_in[19];
    const float* cm_bdt  = (const float*)d_in[20];
    const float* cm_Alog = (const float*)d_in[21];
    const float* cm_Dp   = (const float*)d_in[22];
    const float* cm_Wout = (const float*)d_in[23];
    const float* cm_bout = (const float*)d_in[24];
    float* out = (float*)d_out;

    const long MT = 4096;   // B_*D_  (token-mixer rows)
    const long MC = 1568;   // B_*P_  (channel-mixer rows)

    // ---- workspace carving (256B aligned) ----
    char* base = (char*)d_ws;
    size_t off = 0;
    auto carve = [&](size_t bytes) -> char* {
        char* p = base + off;
        off += (bytes + 255) & ~(size_t)255;
        return p;
    };
    // fp32 intermediates
    float* tm_xz    = (float*)carve(MT * 256 * 4);
    float* tm_xc    = (float*)carve(MT * 128 * 4);
    float* tm_proj  = (float*)carve(MT * 288 * 4);
    float* tm_dtraw = (float*)carve(MT * 128 * 4);
    float* tm_out   = (float*)carve(MT * 224 * 4);
    float* tok_mix  = (float*)carve(MC * 512 * 4);
    float* cm_xz    = (float*)carve(MC * 1024 * 4);
    float* cm_xc    = (float*)carve(MC * 512 * 4);
    float* cm_proj  = (float*)carve(MC * 288 * 4);
    float* cm_dtraw = (float*)carve(MC * 512 * 4);
    // f16 packed operands
    _Float16* t16      = (_Float16*)carve(MT * 224 * 2);   // rmsnorm-T output (pad cols zeroed)
    _Float16* tmxc16   = (_Float16*)carve(MT * 128 * 2);
    _Float16* tmdt16   = (_Float16*)carve(MT * 32 * 2);
    _Float16* tmy16    = (_Float16*)carve(MT * 128 * 2);
    _Float16* cmnorm16 = (_Float16*)carve(MC * 512 * 2);
    _Float16* cmxc16   = (_Float16*)carve(MC * 512 * 2);
    _Float16* cmdt16   = (_Float16*)carve(MC * 32 * 2);
    _Float16* cmy16    = (_Float16*)carve(MC * 512 * 2);
    // f16 transposed padded weights: Wt[Npad, Kp]
    _Float16* tmWin_t  = (_Float16*)carve(256  * 224 * 2);
    _Float16* tmWx_t   = (_Float16*)carve(288  * 128 * 2);
    _Float16* tmWdt_t  = (_Float16*)carve(128  * 32  * 2);
    _Float16* tmWout_t = (_Float16*)carve(224  * 128 * 2);
    _Float16* cmWin_t  = (_Float16*)carve(1024 * 512 * 2);
    _Float16* cmWx_t   = (_Float16*)carve(288  * 512 * 2);
    _Float16* cmWdt_t  = (_Float16*)carve(512  * 32  * 2);
    _Float16* cmWout_t = (_Float16*)carve(512  * 512 * 2);

    #define LAUNCH_EW(kern, total, ...) \
        kern<<<((total) + 255) / 256, 256, 0, stream>>>(__VA_ARGS__)
    #define GEMM_GRID(M, Npad) dim3((Npad) / 32, (M) / 32)

    // ---- pack all weights (transposed, padded, f16) ----
    LAUNCH_EW(pack_weight_kernel, 256 * 224,  tm_Win,  tmWin_t,  196, 256,  224, 256 * 224);
    LAUNCH_EW(pack_weight_kernel, 288 * 128,  tm_Wx,   tmWx_t,   128, 260,  128, 288 * 128);
    LAUNCH_EW(pack_weight_kernel, 128 * 32,   tm_Wdt,  tmWdt_t,  4,   128,  32,  128 * 32);
    LAUNCH_EW(pack_weight_kernel, 224 * 128,  tm_Wout, tmWout_t, 128, 196,  128, 224 * 128);
    LAUNCH_EW(pack_weight_kernel, 1024 * 512, cm_Win,  cmWin_t,  512, 1024, 512, 1024 * 512);
    LAUNCH_EW(pack_weight_kernel, 288 * 512,  cm_Wx,   cmWx_t,   512, 272,  512, 288 * 512);
    LAUNCH_EW(pack_weight_kernel, 512 * 32,   cm_Wdt,  cmWdt_t,  16,  512,  32,  512 * 32);
    LAUNCH_EW(pack_weight_kernel, 512 * 512,  cm_Wout, cmWout_t, 512, 512,  512, 512 * 512);

    // ---- token mixer: sequence axis = D (L=512), features = P (d_in=196) ----
    hipMemsetAsync(t16, 0, (size_t)MT * 224 * 2, stream);   // zero pad cols 196..223
    rmsnorm_kernel<<<B_ * P_, 256, 0, stream>>>(x, tm_norm_w, t16, D_, P_, 224, 1);

    // xz = t @ Win + bin : (4096 x 224f16) x (256 x 224)^T -> (4096,256)
    wmma_gemm_kernel<<<GEMM_GRID(4096, 256), 32, 0, stream>>>(
        t16, tmWin_t, tm_xz, tm_bin, nullptr, 224, 224, 224, 256, 256);

    LAUNCH_EW(conv_silu_kernel, (int)(MT * 128),
              tm_xz, tm_convw, tm_convb, tm_xc, tmxc16, D_, TH_, 256, (int)(MT * 128));

    // proj = xc @ Wx : (4096,128) x (288x128)^T -> (4096,288pad)
    wmma_gemm_kernel<<<GEMM_GRID(4096, 288), 32, 0, stream>>>(
        tmxc16, tmWx_t, tm_proj, nullptr, nullptr, 128, 128, 128, 288, 260);

    LAUNCH_EW(dtslice_kernel, (int)(MT * 32), tm_proj, tmdt16, 288, 4, (int)(MT * 32));

    // dtraw = proj[:,:4] @ Wdt + bdt : (4096,32pad) x (128x32)^T -> (4096,128)
    wmma_gemm_kernel<<<GEMM_GRID(4096, 128), 32, 0, stream>>>(
        tmdt16, tmWdt_t, tm_dtraw, tm_bdt, nullptr, 32, 32, 32, 128, 128);

    mamba_scan_kernel<<<B_ * TH_, 32, 0, stream>>>(
        tm_xc, tm_proj, tm_dtraw, tm_Alog, tm_Dp, tm_xz, tmy16,
        D_, TH_, 4, 288, 256);

    // tm_out = y @ Wout + bout : (4096,128) x (224x128)^T -> (4096,224pad)
    wmma_gemm_kernel<<<GEMM_GRID(4096, 224), 32, 0, stream>>>(
        tmy16, tmWout_t, tm_out, tm_bout, nullptr, 128, 128, 128, 224, 196);

    LAUNCH_EW(transpose_add_kernel, (int)(MC * 512), tm_out, x, tok_mix, (int)(MC * 512));

    // ---- channel mixer: sequence axis = P (L=196), features = D (d_in=512) ----
    rmsnorm_kernel<<<B_ * P_, 256, 0, stream>>>(tok_mix, cm_norm_w, cmnorm16, D_, P_, 0, 0);

    // xz = normed @ Win + bin : (1568,512) x (1024x512)^T -> (1568,1024)
    wmma_gemm_kernel<<<GEMM_GRID(1568, 1024), 32, 0, stream>>>(
        cmnorm16, cmWin_t, cm_xz, cm_bin, nullptr, 512, 512, 512, 1024, 1024);

    LAUNCH_EW(conv_silu_kernel, (int)(MC * 512),
              cm_xz, cm_convw, cm_convb, cm_xc, cmxc16, P_, CH_, 1024, (int)(MC * 512));

    // proj = xc @ Wx : (1568,512) x (288x512)^T -> (1568,288pad)
    wmma_gemm_kernel<<<GEMM_GRID(1568, 288), 32, 0, stream>>>(
        cmxc16, cmWx_t, cm_proj, nullptr, nullptr, 512, 512, 512, 288, 272);

    LAUNCH_EW(dtslice_kernel, (int)(MC * 32), cm_proj, cmdt16, 288, 16, (int)(MC * 32));

    // dtraw = proj[:,:16] @ Wdt + bdt : (1568,32pad) x (512x32)^T -> (1568,512)
    wmma_gemm_kernel<<<GEMM_GRID(1568, 512), 32, 0, stream>>>(
        cmdt16, cmWdt_t, cm_dtraw, cm_bdt, nullptr, 32, 32, 32, 512, 512);

    mamba_scan_kernel<<<B_ * CH_, 32, 0, stream>>>(
        cm_xc, cm_proj, cm_dtraw, cm_Alog, cm_Dp, cm_xz, cmy16,
        P_, CH_, 16, 288, 1024);

    // out = y @ Wout + bout + token_mix : (1568,512) x (512x512)^T -> d_out
    wmma_gemm_kernel<<<GEMM_GRID(1568, 512), 32, 0, stream>>>(
        cmy16, cmWout_t, out, cm_bout, tok_mix, 512, 512, 512, 512, 512);

    #undef LAUNCH_EW
    #undef GEMM_GRID
}
